// CausalPrefixAttention_18064632447488
// MI455X (gfx1250) — compile-verified, matrix operand order checked by reading, not run on hardware
//
#include <hip/hip_runtime.h>
#include <hip/hip_bf16.h>
#include <math.h>

typedef __attribute__((ext_vector_type(16))) __bf16 v16bf;
typedef __attribute__((ext_vector_type(8)))  float  v8f;

#define HEADS 8
#define DIMSZ 1024
#define INNER 512
#define DHEAD 64
#define BATCH 2
#define SEQ   1024
#define JTOT  2048

// ---------------------------------------------------------------------------
// WMMA helpers (v_wmma_f32_16x16x32_bf16, wave32)
// ---------------------------------------------------------------------------
__device__ inline v8f wmma_bf16(v16bf a, v16bf b, v8f c) {
  return __builtin_amdgcn_wmma_f32_16x16x32_bf16(false, a, false, b, (short)0, c,
                                                 false, false);
}

// Load a 16x32 bf16 operand fragment (A row-major, or B via pre-transposed
// [N,K] storage).  ISA 7.12.2 16-bit layout: lane L covers row/col (L&15),
// kbase = (L<16)?0:8; VGPRs 0-3 hold K=kbase+0..7, VGPRs 4-7 hold K=kbase+16..23,
// two bf16 per 32-bit VGPR.  All K offsets even -> aligned 32-bit loads.
__device__ inline v16bf load_frag(const __bf16* __restrict__ base, int row,
                                  int ldk, int k0, int lane) {
  const __bf16* p = base + (size_t)row * ldk + k0 + ((lane & 16) ? 8 : 0);
  const unsigned* pu = (const unsigned*)p;
  union { v16bf v; unsigned u[8]; } f;
  f.u[0] = pu[0];  f.u[1] = pu[1];  f.u[2] = pu[2];  f.u[3] = pu[3];
  f.u[4] = pu[8];  f.u[5] = pu[9];  f.u[6] = pu[10]; f.u[7] = pu[11];
  return f.v;
}

// ---------------------------------------------------------------------------
// Prep kernels
// ---------------------------------------------------------------------------
__global__ void ln_bf16_kernel(const float* __restrict__ x,
                               const float* __restrict__ gamma,
                               const float* __restrict__ beta,
                               __bf16* __restrict__ out) {
  __shared__ float s1[256], s2[256];
  int row = blockIdx.x;
  const float* xr = x + (size_t)row * DIMSZ;
  float a = 0.f, b = 0.f;
  for (int i = threadIdx.x; i < DIMSZ; i += 256) {
    float v = xr[i]; a += v; b += v * v;
  }
  s1[threadIdx.x] = a; s2[threadIdx.x] = b;
  __syncthreads();
  for (int off = 128; off > 0; off >>= 1) {
    if (threadIdx.x < off) {
      s1[threadIdx.x] += s1[threadIdx.x + off];
      s2[threadIdx.x] += s2[threadIdx.x + off];
    }
    __syncthreads();
  }
  float mu   = s1[0] * (1.0f / DIMSZ);
  float var  = s2[0] * (1.0f / DIMSZ) - mu * mu;
  float rstd = rsqrtf(var + 1e-5f);
  __bf16* orow = out + (size_t)row * DIMSZ;
  for (int i = threadIdx.x; i < DIMSZ; i += 256)
    orow[i] = (__bf16)((xr[i] - mu) * rstd * gamma[i] + beta[i]);
}

__global__ void f32_to_bf16_kernel(const float* __restrict__ src,
                                   __bf16* __restrict__ dst, int n) {
  int i = blockIdx.x * 256 + threadIdx.x;
  if (i < n) dst[i] = (__bf16)src[i];
}

// W [K,N] f32 -> Wt [N,K] bf16 (so WMMA B-operand loads are contiguous in K)
__global__ void transpose_bf16_kernel(const float* __restrict__ W,
                                      __bf16* __restrict__ Wt, int K, int N) {
  int idx = blockIdx.x * 256 + threadIdx.x;
  if (idx >= K * N) return;
  int k = idx / N, n = idx % N;
  Wt[(size_t)n * K + k] = (__bf16)W[idx];
}

// ---------------------------------------------------------------------------
// bf16 WMMA GEMM:  C[M,N] = A[M,K] * Bt[N,K]^T
// 256 threads = 8 waves; wave tile 32x32 -> block tile 64x128.
// mode 0: q projection  -> outb = q * 0.125 (bf16, [M,INNER])
// mode 1: kv from xn    (joff=1024) -> K half to kbuf, V half transposed to vt
// mode 2: kv from ctx   (joff=0)    -> same
// mode 3: out projection -> outf = acc + bias (f32, [M,DIMSZ])
// ---------------------------------------------------------------------------
__global__ void __launch_bounds__(256)
gemm_bf16_kernel(const __bf16* __restrict__ A, const __bf16* __restrict__ Bt,
                 int M, int N, int K, int mode, int joff,
                 __bf16* __restrict__ outb, __bf16* __restrict__ vt,
                 float* __restrict__ outf, const float* __restrict__ bias) {
  int lane = threadIdx.x & 31;
  int wave = threadIdx.x >> 5;
  int wm = wave >> 2, wn = wave & 3;
  int m0 = blockIdx.x * 64 + wm * 32;
  int n0 = blockIdx.y * 128 + wn * 32;
  if (m0 >= M || n0 >= N) return;

  int cn = lane & 15;
  v8f acc[2][2] = {};
  for (int k0 = 0; k0 < K; k0 += 32) {
    v16bf a0 = load_frag(A, m0 + cn,      K, k0, lane);
    v16bf a1 = load_frag(A, m0 + 16 + cn, K, k0, lane);
    v16bf b0 = load_frag(Bt, n0 + cn,      K, k0, lane);
    v16bf b1 = load_frag(Bt, n0 + 16 + cn, K, k0, lane);
    acc[0][0] = wmma_bf16(a0, b0, acc[0][0]);
    acc[0][1] = wmma_bf16(a0, b1, acc[0][1]);
    acc[1][0] = wmma_bf16(a1, b0, acc[1][0]);
    acc[1][1] = wmma_bf16(a1, b1, acc[1][1]);
  }

  int rbase = (lane & 16) ? 8 : 0;
  for (int s = 0; s < 2; ++s)
    for (int t = 0; t < 2; ++t)
      for (int r = 0; r < 8; ++r) {
        int m = m0 + s * 16 + rbase + r;
        int n = n0 + t * 16 + cn;
        if (m >= M || n >= N) continue;
        float v = acc[s][t][r];
        if (mode == 0) {
          outb[(size_t)m * INNER + n] = (__bf16)(v * 0.125f);  // d_head^-0.5
        } else if (mode == 3) {
          outf[(size_t)m * DIMSZ + n] = v + bias[n];
        } else {  // kv projection scatter
          int b = m >> 10, i = m & 1023;
          int j = joff + i;
          if (n < INNER) {
            outb[((size_t)b * JTOT + j) * INNER + n] = (__bf16)v;
          } else {
            int nn = n - INNER;
            int h = nn >> 6, d = nn & 63;
            vt[(((size_t)b * HEADS + h) * DHEAD + d) * JTOT + j] = (__bf16)v;
          }
        }
      }
}

// ---------------------------------------------------------------------------
// Flash attention: one wave per (b, h, 16-query tile).
// qb [b,i,h*64+d] bf16 (pre-scaled); kb [b,j,h*64+d] bf16; vt [b,h,d,j] bf16.
// ---------------------------------------------------------------------------
__global__ void __launch_bounds__(256)
attn_kernel(const __bf16* __restrict__ qb, const __bf16* __restrict__ kb,
            const __bf16* __restrict__ vt, __bf16* __restrict__ ao) {
  __shared__ __align__(16) __bf16 pt[8][16][32];  // per-wave prob tile

  int lane = threadIdx.x & 31;
  int wave = threadIdx.x >> 5;
  int w = blockIdx.x * 8 + wave;  // 0..1023 = 2*8*64 wave-tiles
  int b  = w >> 9;
  int h  = (w >> 6) & 7;
  int i0 = (w & 63) * 16;

  const __bf16* qbase = qb + (size_t)b * SEQ * INNER + h * DHEAD;
  const __bf16* kbase = kb + (size_t)b * JTOT * INNER + h * DHEAD;
  const __bf16* vbase = vt + ((size_t)b * HEADS + h) * (size_t)DHEAD * JTOT;

  int cn = lane & 15;
  int rbase = (lane & 16) ? 8 : 0;

  v16bf qf0 = load_frag(qbase, i0 + cn, INNER, 0, lane);
  v16bf qf1 = load_frag(qbase, i0 + cn, INNER, 32, lane);

  float rm[8], rs[8];
#pragma unroll
  for (int r = 0; r < 8; ++r) { rm[r] = -3.0e38f; rs[r] = 0.f; }
  v8f o[4] = {};

  for (int jc = 0; jc < JTOT; jc += 32) {
    // S = q . k^T for 16x32 keys (contraction over d = 64 -> 2 wmma each)
    v8f s0 = {}, s1 = {};
    {
      v16bf k0f = load_frag(kbase, jc + cn,      INNER, 0,  lane);
      v16bf k1f = load_frag(kbase, jc + cn,      INNER, 32, lane);
      s0 = wmma_bf16(qf0, k0f, s0);
      s0 = wmma_bf16(qf1, k1f, s0);
      v16bf k2f = load_frag(kbase, jc + 16 + cn, INNER, 0,  lane);
      v16bf k3f = load_frag(kbase, jc + 16 + cn, INNER, 32, lane);
      s1 = wmma_bf16(qf0, k2f, s1);
      s1 = wmma_bf16(qf1, k3f, s1);
    }
    // mask + online softmax (rows live on 16-lane halves of the wave)
#pragma unroll
    for (int r = 0; r < 8; ++r) {
      int m  = i0 + rbase + r;
      int n0 = jc + cn, n1 = jc + 16 + cn;
      float v0 = s0[r]; if (n0 - 1024 > m) v0 = -3.0e38f;  // causal on input half
      float v1 = s1[r]; if (n1 - 1024 > m) v1 = -3.0e38f;
      float tmax = fmaxf(v0, v1);
      tmax = fmaxf(tmax, __shfl_xor(tmax, 1, 16));
      tmax = fmaxf(tmax, __shfl_xor(tmax, 2, 16));
      tmax = fmaxf(tmax, __shfl_xor(tmax, 4, 16));
      tmax = fmaxf(tmax, __shfl_xor(tmax, 8, 16));
      float nm = fmaxf(rm[r], tmax);
      float alpha = __expf(rm[r] - nm);
      rm[r] = nm;
      float p0 = __expf(v0 - nm);
      float p1 = __expf(v1 - nm);
      float ps = p0 + p1;
      ps += __shfl_xor(ps, 1, 16);
      ps += __shfl_xor(ps, 2, 16);
      ps += __shfl_xor(ps, 4, 16);
      ps += __shfl_xor(ps, 8, 16);
      rs[r] = rs[r] * alpha + ps;
      o[0][r] *= alpha; o[1][r] *= alpha; o[2][r] *= alpha; o[3][r] *= alpha;
      pt[wave][rbase + r][cn]      = (__bf16)p0;
      pt[wave][rbase + r][cn + 16] = (__bf16)p1;
    }
    __syncthreads();  // uniform trip count across all 8 waves
    // O += P . V   (contraction over 32 keys; B operand from transposed V)
    v16bf pf = load_frag(&pt[wave][0][0], cn, 32, 0, lane);
#pragma unroll
    for (int t = 0; t < 4; ++t) {
      v16bf vf = load_frag(vbase, t * 16 + cn, JTOT, jc, lane);
      o[t] = wmma_bf16(pf, vf, o[t]);
    }
    __syncthreads();
  }

  __bf16* arow = ao + (size_t)b * SEQ * INNER + h * DHEAD;
#pragma unroll
  for (int r = 0; r < 8; ++r) {
    float inv = 1.0f / rs[r];
    int m = i0 + rbase + r;
#pragma unroll
    for (int t = 0; t < 4; ++t)
      arow[(size_t)m * INNER + t * 16 + cn] = (__bf16)(o[t][r] * inv);
  }
}

// ---------------------------------------------------------------------------
// Launch
// ---------------------------------------------------------------------------
extern "C" void kernel_launch(void* const* d_in, const int* in_sizes, int n_in,
                              void* d_out, int out_size, void* d_ws,
                              size_t ws_size, hipStream_t stream) {
  const float* x     = (const float*)d_in[0];
  const float* ctx   = (const float*)d_in[1];
  const float* gamma = (const float*)d_in[2];
  const float* beta  = (const float*)d_in[3];
  const float* Wq    = (const float*)d_in[4];
  const float* Wkv   = (const float*)d_in[5];
  const float* Wo    = (const float*)d_in[6];
  const float* bo    = (const float*)d_in[7];
  float* out = (float*)d_out;

  char* p = (char*)d_ws;
  auto alloc = [&](size_t bytes) -> char* {
    char* r = p;
    p += (bytes + 255) & ~(size_t)255;
    return r;
  };
  __bf16* xn   = (__bf16*)alloc((size_t)BATCH * SEQ * DIMSZ * 2);
  __bf16* ctxb = (__bf16*)alloc((size_t)BATCH * SEQ * DIMSZ * 2);
  __bf16* Wqt  = (__bf16*)alloc((size_t)INNER * DIMSZ * 2);
  __bf16* Wkvt = (__bf16*)alloc((size_t)2 * INNER * DIMSZ * 2);
  __bf16* Wot  = (__bf16*)alloc((size_t)DIMSZ * INNER * 2);
  __bf16* qb   = (__bf16*)alloc((size_t)BATCH * SEQ * INNER * 2);
  __bf16* kbuf = (__bf16*)alloc((size_t)BATCH * JTOT * INNER * 2);
  __bf16* vtb  = (__bf16*)alloc((size_t)BATCH * HEADS * DHEAD * JTOT * 2);
  __bf16* aob  = (__bf16*)alloc((size_t)BATCH * SEQ * INNER * 2);
  (void)ws_size; (void)in_sizes; (void)n_in; (void)out_size;

  // Precision prep (one-shot, deterministic each call)
  ln_bf16_kernel<<<BATCH * SEQ, 256, 0, stream>>>(x, gamma, beta, xn);
  int nctx = BATCH * SEQ * DIMSZ;
  f32_to_bf16_kernel<<<(nctx + 255) / 256, 256, 0, stream>>>(ctx, ctxb, nctx);
  transpose_bf16_kernel<<<(DIMSZ * INNER + 255) / 256, 256, 0, stream>>>(
      Wq, Wqt, DIMSZ, INNER);
  transpose_bf16_kernel<<<(DIMSZ * 2 * INNER + 255) / 256, 256, 0, stream>>>(
      Wkv, Wkvt, DIMSZ, 2 * INNER);
  transpose_bf16_kernel<<<(INNER * DIMSZ + 255) / 256, 256, 0, stream>>>(
      Wo, Wot, INNER, DIMSZ);

  // q = xn.Wq * 0.125
  gemm_bf16_kernel<<<dim3(2048 / 64, 512 / 128), 256, 0, stream>>>(
      xn, Wqt, 2048, 512, 1024, 0, 0, qb, nullptr, nullptr, nullptr);
  // kv_in = xn.Wkv  (input keys go at j = 1024 + i)
  gemm_bf16_kernel<<<dim3(2048 / 64, 1024 / 128), 256, 0, stream>>>(
      xn, Wkvt, 2048, 1024, 1024, 1, 1024, kbuf, vtb, nullptr, nullptr);
  // kv_cx = ctx.Wkv (context keys at j = i)
  gemm_bf16_kernel<<<dim3(2048 / 64, 1024 / 128), 256, 0, stream>>>(
      ctxb, Wkvt, 2048, 1024, 1024, 2, 0, kbuf, vtb, nullptr, nullptr);
  // flash attention
  attn_kernel<<<128, 256, 0, stream>>>(qb, kbuf, vtb, aob);
  // out = ao.Wo + bo
  gemm_bf16_kernel<<<dim3(2048 / 64, 1024 / 128), 256, 0, stream>>>(
      aob, Wot, 2048, 1024, 512, 3, 0, nullptr, nullptr, out, bo);
}
